// MultiHeadAttentionBlock_1872605741104
// MI455X (gfx1250) — matrix operand order repaired and checked by reading, not compiled, not run on hardware
//
#include <hip/hip_runtime.h>
#include <hip/hip_bf16.h>

// ---------------------------------------------------------------------------
// Multi-head attention block for MI455X (gfx1250), bf16 WMMA + fp32 accumulate
// B=2, S=2048, D=1024, H=16, DK=64
// Data movement: GLOBAL_LOAD_ASYNC_TO_LDS_B128 (ASYNCcnt) for bf16 tiles,
// VGPR-path loads only where conversion/transpose is required.
// ---------------------------------------------------------------------------

#define BQ   2
#define SQ   2048
#define DM   1024
#define NH   16
#define DKH  64
#define MROWS (BQ * SQ)          // 4096

typedef __attribute__((ext_vector_type(16))) __bf16 v16bf;
typedef __attribute__((ext_vector_type(8)))  float  v8f;

union ABFrag { v16bf v; unsigned u[8]; };
union CFrag  { v8f   v; float    f[8]; };

static __device__ __forceinline__ unsigned short f2bf(float x) {
  unsigned u = __float_as_uint(x);
  u += 0x7FFFu + ((u >> 16) & 1u);        // round-to-nearest-even
  return (unsigned short)(u >> 16);
}
static __device__ __forceinline__ unsigned pk2(float a, float b) {
  return (unsigned)f2bf(a) | ((unsigned)f2bf(b) << 16);
}
static __device__ __forceinline__ float rowmax16(float v) {
#pragma unroll
  for (int m = 1; m < 16; m <<= 1) v = fmaxf(v, __shfl_xor(v, m, 32));
  return v;
}
static __device__ __forceinline__ float rowsum16(float v) {
#pragma unroll
  for (int m = 1; m < 16; m <<= 1) v += __shfl_xor(v, m, 32);
  return v;
}

// Async memory->LDS copy of 16B per lane (GV mode, ASYNCcnt-tracked).
// lds_off: byte offset in LDS (flat shared addr truncates to LDS_ADDR per ISA 10.2)
static __device__ __forceinline__ void async_copy_b128(void* lds_ptr, const void* gptr) {
  unsigned lds_off = (unsigned)(size_t)lds_ptr;
  unsigned long long ga = (unsigned long long)(size_t)gptr;
  asm volatile("global_load_async_to_lds_b128 %0, %1, off"
               :: "v"(lds_off), "v"(ga) : "memory");
}
static __device__ __forceinline__ void wait_async0() {
#if __has_builtin(__builtin_amdgcn_s_wait_asynccnt)
  __builtin_amdgcn_s_wait_asynccnt(0);
#else
  asm volatile("s_wait_asynccnt 0x0" ::: "memory");
#endif
}

// ---------------------------------------------------------------------------
// GEMM tile config (shared by projection kernels)
// ---------------------------------------------------------------------------
#define GBM 128
#define GBN 64
#define GBK 32
#define LPAD 8
#define LSTR (GBK + LPAD)        // 40 ushorts per LDS row (80B: 16B aligned rows)

// ---------------------------------------------------------------------------
// QKV projection: Out_bf16[B,H,S,DK] = X_f32[B*S,D] @ W_f32[D,D] + bias
// (fp32 sources need conversion, so tiles go through the VGPR path)
// ---------------------------------------------------------------------------
__global__ __launch_bounds__(256)
void qkv_gemm_kernel(const float* __restrict__ X, const float* __restrict__ W,
                     const float* __restrict__ bias, unsigned short* __restrict__ Out) {
  __shared__ unsigned short Alds[GBM * LSTR];
  __shared__ unsigned short Blds[GBN * LSTR];

  const int t    = threadIdx.x;
  const int lane = t & 31, wave = t >> 5;
  const int wm   = wave >> 1, wn = wave & 1;
  const int m0   = blockIdx.x * GBM, n0 = blockIdx.y * GBN;
  const int lm   = lane & 15, kh = lane >> 4;

  CFrag acc[2][2];
#pragma unroll
  for (int i = 0; i < 2; i++)
#pragma unroll
    for (int j = 0; j < 2; j++)
#pragma unroll
      for (int r = 0; r < 8; r++) acc[i][j].f[r] = 0.0f;

  for (int k0 = 0; k0 < DM; k0 += GBK) {
    // A tile 128x32 fp32 -> bf16 LDS (row-major, padded)
#pragma unroll
    for (int i = 0; i < 4; i++) {
      int row = (t >> 3) + i * 32;
      int col = (t & 7) * 4;
      const float4 a = *reinterpret_cast<const float4*>(X + (size_t)(m0 + row) * DM + k0 + col);
      unsigned* dst = reinterpret_cast<unsigned*>(&Alds[row * LSTR + col]);
      dst[0] = pk2(a.x, a.y);
      dst[1] = pk2(a.z, a.w);
    }
    // W tile 32x64 fp32 -> transposed bf16 LDS [n][k] (contiguous along k)
#pragma unroll
    for (int i = 0; i < 4; i++) {
      int unit = t * 4 + i;          // 1024 units = 64 n * 16 k-pairs
      int n  = unit >> 4;
      int kp = unit & 15;
      float w0 = W[(size_t)(k0 + 2 * kp)     * DM + n0 + n];
      float w1 = W[(size_t)(k0 + 2 * kp + 1) * DM + n0 + n];
      *reinterpret_cast<unsigned*>(&Blds[n * LSTR + 2 * kp]) = pk2(w0, w1);
    }
    __syncthreads();

    ABFrag afr[2], bfr[2];
#pragma unroll
    for (int tm = 0; tm < 2; tm++) {
      int mrow = (wm * 32 + tm * 16 + lm) * LSTR;
#pragma unroll
      for (int j = 0; j < 8; j++) {
        int k = ((j >> 2) << 4) + kh * 8 + ((j & 3) << 1);   // A 16-bit layout
        afr[tm].u[j] = *reinterpret_cast<const unsigned*>(&Alds[mrow + k]);
      }
    }
#pragma unroll
    for (int tn = 0; tn < 2; tn++) {
      int nrow = (wn * 32 + tn * 16 + lm) * LSTR;
#pragma unroll
      for (int j = 0; j < 8; j++) {
        int k = ((j >> 2) << 4) + kh * 8 + ((j & 3) << 1);   // B 16-bit layout
        bfr[tn].u[j] = *reinterpret_cast<const unsigned*>(&Blds[nrow + k]);
      }
    }
#pragma unroll
    for (int tm = 0; tm < 2; tm++)
#pragma unroll
      for (int tn = 0; tn < 2; tn++)
        acc[tm][tn].v = __builtin_amdgcn_wmma_f32_16x16x32_bf16(
            false, afr[tm].v, false, bfr[tn].v, (short)0, acc[tm][tn].v, false, false);
    __syncthreads();
  }

  // write bf16 output in [B,H,S,DK] layout, adding bias
#pragma unroll
  for (int tm = 0; tm < 2; tm++) {
#pragma unroll
    for (int tn = 0; tn < 2; tn++) {
      int ng = n0 + wn * 32 + tn * 16 + lm;
      int h = ng >> 6, dk = ng & 63;
      float bvv = bias[ng];
#pragma unroll
      for (int r = 0; r < 8; r++) {
        int mg = m0 + wm * 32 + tm * 16 + r + kh * 8;
        int b = mg >> 11, s = mg & 2047;
        Out[((size_t)(b * NH + h) * SQ + s) * DKH + dk] = f2bf(acc[tm][tn].f[r] + bvv);
      }
    }
  }
}

// ---------------------------------------------------------------------------
// Causal flash attention: Ctx_bf16[B,S,D] from Q/K/V bf16 [B,H,S,DK]
// One block = 128 query rows of one (b,h); 8 waves x 16 rows each.
// K tile: async memory->LDS copy (no VGPR round trip).
// V tile: VGPR path (needs transpose into [dk][kv]).
// ---------------------------------------------------------------------------
#define AP 72   // padded 64-wide LDS rows (144B: 16B aligned)

__global__ __launch_bounds__(256)
void flash_attn_kernel(const unsigned short* __restrict__ Q,
                       const unsigned short* __restrict__ K,
                       const unsigned short* __restrict__ V,
                       unsigned short* __restrict__ Ctx) {
  __shared__ unsigned short Klds[64 * AP];        // [kv][dk]
  __shared__ unsigned short Vlds[64 * AP];        // [dk][kv] (transposed)
  __shared__ unsigned short Plds[8 * 16 * AP];    // per-wave P scratch

  const int t    = threadIdx.x;
  const int lane = t & 31, wave = t >> 5;
  const int lm   = lane & 15, kh = lane >> 4;
  const int bh   = blockIdx.y;
  const int q0   = blockIdx.x * 128 + wave * 16;
  const size_t head = (size_t)bh * SQ * DKH;

  // Preload Q A-fragments (2 K-steps of 32)
  ABFrag qfr[2];
#pragma unroll
  for (int s = 0; s < 2; s++)
#pragma unroll
    for (int j = 0; j < 8; j++) {
      int k = s * 32 + ((j >> 2) << 4) + kh * 8 + ((j & 3) << 1);
      qfr[s].u[j] = *reinterpret_cast<const unsigned*>(Q + head + (size_t)(q0 + lm) * DKH + k);
    }

  CFrag acc[4];
#pragma unroll
  for (int d = 0; d < 4; d++)
#pragma unroll
    for (int r = 0; r < 8; r++) acc[d].f[r] = 0.0f;
  float mrun[8], lrun[8];
#pragma unroll
  for (int r = 0; r < 8; r++) { mrun[r] = -3.0e38f; lrun[r] = 0.0f; }

  const int nkv = (blockIdx.x + 1) * 2;          // causal: kv < (qt+1)*128
  unsigned short* Pw = &Plds[wave * 16 * AP];

  for (int kt = 0; kt < nkv; kt++) {
    const int kv0 = kt * 64;
    // K tile: async copy straight into LDS natural layout (ASYNCcnt path)
#pragma unroll
    for (int i = 0; i < 2; i++) {
      int row = (t >> 3) + i * 32;
      int off = (t & 7) * 8;
      async_copy_b128(&Klds[row * AP + off],
                      K + head + (size_t)(kv0 + row) * DKH + off);
    }
    // V tile: VGPR path with transpose into Vlds[dk][kv]
#pragma unroll
    for (int i = 0; i < 2; i++) {
      int row = (t >> 3) + i * 32;
      int off = (t & 7) * 8;
      const uint4 vq = *reinterpret_cast<const uint4*>(V + head + (size_t)(kv0 + row) * DKH + off);
      Vlds[(off + 0) * AP + row] = (unsigned short)(vq.x);
      Vlds[(off + 1) * AP + row] = (unsigned short)(vq.x >> 16);
      Vlds[(off + 2) * AP + row] = (unsigned short)(vq.y);
      Vlds[(off + 3) * AP + row] = (unsigned short)(vq.y >> 16);
      Vlds[(off + 4) * AP + row] = (unsigned short)(vq.z);
      Vlds[(off + 5) * AP + row] = (unsigned short)(vq.z >> 16);
      Vlds[(off + 6) * AP + row] = (unsigned short)(vq.w);
      Vlds[(off + 7) * AP + row] = (unsigned short)(vq.w >> 16);
    }
    // prefetch next KV tile into cache while this one is consumed
    if (kt + 1 < nkv) {
      int row = (t >> 3);
      __builtin_prefetch(K + head + (size_t)(kv0 + 64 + row) * DKH, 0, 3);
      __builtin_prefetch(V + head + (size_t)(kv0 + 64 + row) * DKH, 0, 3);
    }
    wait_async0();          // our wave's async K-tile writes complete
    __syncthreads();        // all waves' K/V tile writes visible

    if (kv0 <= q0 + 15) {   // wave-uniform skip of fully-masked tiles (EXEC stays all-1s)
      CFrag sc[4];
#pragma unroll
      for (int n = 0; n < 4; n++)
#pragma unroll
        for (int r = 0; r < 8; r++) sc[n].f[r] = 0.0f;

      // scores = Q @ K^T  (B frag: lane n = kv, k = dk -> Klds rows contiguous)
#pragma unroll
      for (int s = 0; s < 2; s++) {
#pragma unroll
        for (int n = 0; n < 4; n++) {
          ABFrag kfr;
#pragma unroll
          for (int j = 0; j < 8; j++) {
            int k = s * 32 + ((j >> 2) << 4) + kh * 8 + ((j & 3) << 1);
            kfr.u[j] = *reinterpret_cast<const unsigned*>(&Klds[(n * 16 + lm) * AP + k]);
          }
          sc[n].v = __builtin_amdgcn_wmma_f32_16x16x32_bf16(
              false, qfr[s].v, false, kfr.v, (short)0, sc[n].v, false, false);
        }
      }
      // scale + causal mask
#pragma unroll
      for (int n = 0; n < 4; n++) {
        int kvg = kv0 + n * 16 + lm;
#pragma unroll
        for (int r = 0; r < 8; r++) {
          float vv = sc[n].f[r] * 0.125f;             // 1/sqrt(64)
          int qg = q0 + r + kh * 8;
          sc[n].f[r] = (kvg > qg) ? -1.0e30f : vv;
        }
      }
      // online softmax (row groups live in 16-lane halves of the C layout)
#pragma unroll
      for (int r = 0; r < 8; r++) {
        float mx = fmaxf(fmaxf(sc[0].f[r], sc[1].f[r]), fmaxf(sc[2].f[r], sc[3].f[r]));
        mx = rowmax16(mx);
        float nm = fmaxf(mrun[r], mx);
        float alpha = __expf(mrun[r] - nm);
        mrun[r] = nm;
        float sum = 0.0f;
#pragma unroll
        for (int n = 0; n < 4; n++) {
          float p = __expf(sc[n].f[r] - nm);
          sc[n].f[r] = p;
          sum += p;
        }
        sum = rowsum16(sum);
        lrun[r] = lrun[r] * alpha + sum;
#pragma unroll
        for (int d = 0; d < 4; d++) acc[d].f[r] *= alpha;
      }
      // P: C layout -> bf16 A layout via per-wave LDS scratch
#pragma unroll
      for (int n = 0; n < 4; n++)
#pragma unroll
        for (int r = 0; r < 8; r++)
          Pw[(r + kh * 8) * AP + n * 16 + lm] = f2bf(sc[n].f[r]);

      // ctx += P @ V  (B frag: lane n = dk, k = kv -> transposed Vlds contiguous)
#pragma unroll
      for (int s = 0; s < 2; s++) {
        ABFrag pfr;
#pragma unroll
        for (int j = 0; j < 8; j++) {
          int k = s * 32 + ((j >> 2) << 4) + kh * 8 + ((j & 3) << 1);
          pfr.u[j] = *reinterpret_cast<const unsigned*>(&Pw[lm * AP + k]);
        }
#pragma unroll
        for (int d = 0; d < 4; d++) {
          ABFrag vfr;
#pragma unroll
          for (int j = 0; j < 8; j++) {
            int k = s * 32 + ((j >> 2) << 4) + kh * 8 + ((j & 3) << 1);
            vfr.u[j] = *reinterpret_cast<const unsigned*>(&Vlds[(d * 16 + lm) * AP + k]);
          }
          acc[d].v = __builtin_amdgcn_wmma_f32_16x16x32_bf16(
              false, pfr.v, false, vfr.v, (short)0, acc[d].v, false, false);
        }
      }
    }
    __syncthreads();
  }

  // normalize + write ctx as bf16 in [B,S,D] layout
  const int b = bh >> 4, h = bh & 15;
#pragma unroll
  for (int d = 0; d < 4; d++) {
#pragma unroll
    for (int r = 0; r < 8; r++) {
      int s = q0 + r + kh * 8;
      float vv = acc[d].f[r] / lrun[r];
      Ctx[((size_t)b * SQ + s) * DM + h * DKH + d * 16 + lm] = f2bf(vv);
    }
  }
}

// ---------------------------------------------------------------------------
// Output projection: out_f32[B*S,D] = ctx_bf16[B*S,D] @ Wo + bo
// A tile is bf16 already -> async memory->LDS copy.
// ---------------------------------------------------------------------------
__global__ __launch_bounds__(256)
void out_gemm_kernel(const unsigned short* __restrict__ Xbf, const float* __restrict__ W,
                     const float* __restrict__ bias, float* __restrict__ Out) {
  __shared__ unsigned short Alds[GBM * LSTR];
  __shared__ unsigned short Blds[GBN * LSTR];

  const int t    = threadIdx.x;
  const int lane = t & 31, wave = t >> 5;
  const int wm   = wave >> 1, wn = wave & 1;
  const int m0   = blockIdx.x * GBM, n0 = blockIdx.y * GBN;
  const int lm   = lane & 15, kh = lane >> 4;

  CFrag acc[2][2];
#pragma unroll
  for (int i = 0; i < 2; i++)
#pragma unroll
    for (int j = 0; j < 2; j++)
#pragma unroll
      for (int r = 0; r < 8; r++) acc[i][j].f[r] = 0.0f;

  for (int k0 = 0; k0 < DM; k0 += GBK) {
    // A tile: bf16 source -> async copy into padded LDS (16B rows stay aligned)
#pragma unroll
    for (int i = 0; i < 2; i++) {
      int idx = t + i * 256;           // 512 x 16B covers 128x32 bf16
      int row = idx >> 2;
      int qo  = (idx & 3) * 8;
      async_copy_b128(&Alds[row * LSTR + qo],
                      Xbf + (size_t)(m0 + row) * DM + k0 + qo);
    }
#pragma unroll
    for (int i = 0; i < 4; i++) {
      int unit = t * 4 + i;
      int n  = unit >> 4;
      int kp = unit & 15;
      float w0 = W[(size_t)(k0 + 2 * kp)     * DM + n0 + n];
      float w1 = W[(size_t)(k0 + 2 * kp + 1) * DM + n0 + n];
      *reinterpret_cast<unsigned*>(&Blds[n * LSTR + 2 * kp]) = pk2(w0, w1);
    }
    wait_async0();
    __syncthreads();

    ABFrag afr[2], bfr[2];
#pragma unroll
    for (int tm = 0; tm < 2; tm++) {
      int mrow = (wm * 32 + tm * 16 + lm) * LSTR;
#pragma unroll
      for (int j = 0; j < 8; j++) {
        int k = ((j >> 2) << 4) + kh * 8 + ((j & 3) << 1);
        afr[tm].u[j] = *reinterpret_cast<const unsigned*>(&Alds[mrow + k]);
      }
    }
#pragma unroll
    for (int tn = 0; tn < 2; tn++) {
      int nrow = (wn * 32 + tn * 16 + lm) * LSTR;
#pragma unroll
      for (int j = 0; j < 8; j++) {
        int k = ((j >> 2) << 4) + kh * 8 + ((j & 3) << 1);
        bfr[tn].u[j] = *reinterpret_cast<const unsigned*>(&Blds[nrow + k]);
      }
    }
#pragma unroll
    for (int tm = 0; tm < 2; tm++)
#pragma unroll
      for (int tn = 0; tn < 2; tn++)
        acc[tm][tn].v = __builtin_amdgcn_wmma_f32_16x16x32_bf16(
            false, afr[tm].v, false, bfr[tn].v, (short)0, acc[tm][tn].v, false, false);
    __syncthreads();
  }

#pragma unroll
  for (int tm = 0; tm < 2; tm++) {
#pragma unroll
    for (int tn = 0; tn < 2; tn++) {
      int ng = n0 + wn * 32 + tn * 16 + lm;
      float bvv = bias[ng];
#pragma unroll
      for (int r = 0; r < 8; r++) {
        int mg = m0 + wm * 32 + tm * 16 + r + kh * 8;
        Out[(size_t)mg * DM + ng] = acc[tm][tn].f[r] + bvv;
      }
    }
  }
}

// ---------------------------------------------------------------------------
extern "C" void kernel_launch(void* const* d_in, const int* in_sizes, int n_in,
                              void* d_out, int out_size, void* d_ws, size_t ws_size,
                              hipStream_t stream) {
  (void)in_sizes; (void)n_in; (void)out_size; (void)ws_size;
  const float* q  = (const float*)d_in[0];
  const float* k  = (const float*)d_in[1];
  const float* v  = (const float*)d_in[2];
  // d_in[3] is the causal mask (tril) — applied analytically in-kernel.
  const float* Wq = (const float*)d_in[4];
  const float* bq = (const float*)d_in[5];
  const float* Wk = (const float*)d_in[6];
  const float* bk = (const float*)d_in[7];
  const float* Wv = (const float*)d_in[8];
  const float* bv = (const float*)d_in[9];
  const float* Wo = (const float*)d_in[10];
  const float* bo = (const float*)d_in[11];
  float* out = (float*)d_out;

  // workspace: Q, K, V ([B,H,S,DK] bf16) + ctx ([B,S,D] bf16) = 32 MB
  unsigned short* Qws = (unsigned short*)d_ws;
  unsigned short* Kws = Qws + (size_t)MROWS * DM;
  unsigned short* Vws = Kws + (size_t)MROWS * DM;
  unsigned short* Cws = Vws + (size_t)MROWS * DM;

  dim3 blk(256);
  dim3 ggemm(MROWS / GBM, DM / GBN);       // 32 x 16
  qkv_gemm_kernel<<<ggemm, blk, 0, stream>>>(q, Wq, bq, Qws);
  qkv_gemm_kernel<<<ggemm, blk, 0, stream>>>(k, Wk, bk, Kws);
  qkv_gemm_kernel<<<ggemm, blk, 0, stream>>>(v, Wv, bv, Vws);

  dim3 gattn(SQ / 128, BQ * NH);           // 16 x 32
  flash_attn_kernel<<<gattn, blk, 0, stream>>>(Qws, Kws, Vws, Cws);

  out_gemm_kernel<<<ggemm, blk, 0, stream>>>(Cws, Wo, bo, out);
}
